// CNN_binary_69260642615466
// MI455X (gfx1250) — 8-bit matrix operand order repaired and checked by reading, compiled, not run on hardware
//
#include <hip/hip_runtime.h>

typedef int v8i __attribute__((ext_vector_type(8)));

#define BN_EPS 1e-5f

__device__ __forceinline__ float sgnf(float x) { return x >= 0.f ? 1.f : -1.f; }
__device__ __forceinline__ int   sgnb(float x) { return x >= 0.f ? 1 : 0xFF; }

__device__ __forceinline__ int pack4(int b0, int b1, int b2, int b3) {
    return (b0 & 0xFF) | ((b1 & 0xFF) << 8) | ((b2 & 0xFF) << 16) | ((b3 & 0xFF) << 24);
}

__device__ __forceinline__ v8i frag_a(const char* p, int lane) {
    p += (lane & 16) ? 8 : 0;
    v8i a;
    a[0] = *(const int*)(p +  0); a[1] = *(const int*)(p +  4);
    a[2] = *(const int*)(p + 16); a[3] = *(const int*)(p + 20);
    a[4] = *(const int*)(p + 32); a[5] = *(const int*)(p + 36);
    a[6] = *(const int*)(p + 48); a[7] = *(const int*)(p + 52);
    return a;
}
__device__ __forceinline__ v8i frag_b(const char* p, int lane) {
    p += (lane & 16) ? 8 : 0;
    v8i b;
    b[0] = *(const int*)(p +  0); b[1] = *(const int*)(p +  4);
    b[2] = *(const int*)(p + 16); b[3] = *(const int*)(p + 20);
    b[4] = *(const int*)(p + 32); b[5] = *(const int*)(p + 36);
    b[6] = *(const int*)(p + 48); b[7] = *(const int*)(p + 52);
    return b;
}

__global__ __launch_bounds__(256) void k_block1(
    const float* __restrict__ x, const float* __restrict__ w1, const float* __restrict__ b1,
    const float* __restrict__ g1, const float* __restrict__ be1,
    const float* __restrict__ m1, const float* __restrict__ v1, char* __restrict__ A1)
{
    __shared__ float ws[288];
    __shared__ float inv[32], beta[32], bia[32];
    int t = threadIdx.x;
    for (int i = t; i < 288; i += 256) ws[i] = sgnf(w1[i]);
    if (t < 32) {
        float iv = g1[t] * rsqrtf(v1[t] + BN_EPS);
        inv[t] = iv; beta[t] = be1[t] - m1[t] * iv; bia[t] = b1[t];
    }
    __syncthreads();
    long long gid = (long long)blockIdx.x * 256 + t;
    int oc = (int)(gid & 31);
    long long s = gid >> 5;
    int wp = (int)(s & 31);
    long long s2 = s >> 5;
    int h = (int)(s2 % 6);
    int b = (int)(s2 / 6);
    const float* xr = x + (long long)(b * 6 + h) * 128;
    float best = -1e30f;
#pragma unroll
    for (int j = 0; j < 2; ++j) {
        int iw0 = (wp * 2 + j) * 2 - 4;
        float acc = 0.f;
#pragma unroll
        for (int kw = 0; kw < 9; ++kw) {
            int iw = iw0 + kw;
            if (iw >= 0 && iw < 128) acc += sgnf(xr[iw]) * ws[oc * 9 + kw];
        }
        float y = (acc + bia[oc]) * inv[oc] + beta[oc];
        best = fmaxf(best, y);
    }
    A1[gid] = best >= 0.f ? (char)1 : (char)-1;
}

__global__ __launch_bounds__(256) void k_block2(
    const char* __restrict__ A1, const float* __restrict__ w2, const float* __restrict__ b2,
    const float* __restrict__ g2, const float* __restrict__ be2,
    const float* __restrict__ m2, const float* __restrict__ v2, char* __restrict__ A2)
{
    __shared__ char As[64 * 128];
    __shared__ char Bs[128 * 128];
    int t = threadIdx.x;
    int Nbase = blockIdx.x * 128;
    for (int d = t; d < 2048; d += 256) {
        int m = d >> 5, k4 = (d & 31) << 2, out = 0;
        if (k4 < 96) {
            int kw = k4 >> 5, ic = k4 & 31;
            out = pack4(sgnb(w2[m*96 + (ic  )*3 + kw]), sgnb(w2[m*96 + (ic+1)*3 + kw]),
                        sgnb(w2[m*96 + (ic+2)*3 + kw]), sgnb(w2[m*96 + (ic+3)*3 + kw]));
        }
        *(int*)&As[d << 2] = out;
    }
    for (int d = t; d < 4096; d += 256) {
        int n = d >> 5, k4 = (d & 31) << 2, val = 0;
        int N = Nbase + n, b = N / 192, rem = N % 192, h = rem >> 5, w = rem & 31;
        if (k4 < 96) {
            int kw = k4 >> 5, ic = k4 & 31, wi = w + kw - 1;
            if (wi >= 0 && wi < 32)
                val = *(const int*)&A1[(((b*6 + h) * 32 + wi) << 5) + ic];
        }
        *(int*)&Bs[d << 2] = val;
    }
    __syncthreads();

    int lane = t & 31, wv = t >> 5, n0 = wv << 4;
    int nr = n0 + (lane & 15), mr = lane & 15;
    v8i acc[4]; for (int i = 0; i < 4; ++i) acc[i] = (v8i){};
#pragma unroll
    for (int kc = 0; kc < 2; ++kc) {
        v8i bf = frag_b(&Bs[nr * 128 + kc * 64], lane);
#pragma unroll
        for (int mt = 0; mt < 4; ++mt) {
            v8i af = frag_a(&As[(mt * 16 + mr) * 128 + kc * 64], lane);
            acc[mt] = __builtin_amdgcn_wmma_i32_16x16x64_iu8(true, af, true, bf, acc[mt], false, false);
        }
    }
    int N = Nbase + nr, b = N / 192, rem = N % 192, h = rem >> 5, w = rem & 31;
    char* outp = &A2[((b*6 + h) * 32 + w) << 6];
#pragma unroll
    for (int mt = 0; mt < 4; ++mt)
#pragma unroll
        for (int r = 0; r < 8; ++r) {
            int m = mt * 16 + r + ((lane >> 4) << 3);
            float iv = g2[m] * rsqrtf(v2[m] + BN_EPS);
            float val = ((float)acc[mt][r] + b2[m]) * iv + (be2[m] - m2[m] * iv);
            outp[m] = val >= 0.f ? (char)1 : (char)-1;
        }
}

__global__ __launch_bounds__(256) void k_block3(
    const char* __restrict__ A2, const float* __restrict__ w3, const float* __restrict__ b3,
    const float* __restrict__ g3, const float* __restrict__ be3,
    const float* __restrict__ m3, const float* __restrict__ v3, char* __restrict__ A3)
{
    __shared__ char As[128 * 192];
    __shared__ char Bs[128 * 192];
    int t = threadIdx.x;
    int Nbase = blockIdx.x * 128;
    for (int d = t; d < 6144; d += 256) {
        int m = d / 48, k4 = (d % 48) << 2;
        int kw = k4 >> 6, ic = k4 & 63;
        *(int*)&As[d << 2] =
            pack4(sgnb(w3[m*192 + (ic  )*3 + kw]), sgnb(w3[m*192 + (ic+1)*3 + kw]),
                  sgnb(w3[m*192 + (ic+2)*3 + kw]), sgnb(w3[m*192 + (ic+3)*3 + kw]));
    }
    for (int d = t; d < 6144; d += 256) {
        int n = d / 48, k4 = (d % 48) << 2, val = 0;
        int N = Nbase + n, b = N / 192, rem = N % 192, h = rem >> 5, w = rem & 31;
        int kw = k4 >> 6, ic = k4 & 63, wi = w + kw - 1;
        if (wi >= 0 && wi < 32)
            val = *(const int*)&A2[(((b*6 + h) * 32 + wi) << 6) + ic];
        *(int*)&Bs[d << 2] = val;
    }
    __syncthreads();

    int lane = t & 31, wv = t >> 5, n0 = wv << 4;
    int nr = n0 + (lane & 15), mr = lane & 15;
    v8i acc[8]; for (int i = 0; i < 8; ++i) acc[i] = (v8i){};
#pragma unroll
    for (int kc = 0; kc < 3; ++kc) {
        v8i bf = frag_b(&Bs[nr * 192 + kc * 64], lane);
#pragma unroll
        for (int mt = 0; mt < 8; ++mt) {
            v8i af = frag_a(&As[(mt * 16 + mr) * 192 + kc * 64], lane);
            acc[mt] = __builtin_amdgcn_wmma_i32_16x16x64_iu8(true, af, true, bf, acc[mt], false, false);
        }
    }
    int N = Nbase + nr, b = N / 192, rem = N % 192, h = rem >> 5, w = rem & 31;
#pragma unroll
    for (int mt = 0; mt < 8; ++mt)
#pragma unroll
        for (int r = 0; r < 8; ++r) {
            int m = mt * 16 + r + ((lane >> 4) << 3);
            float iv = g3[m] * rsqrtf(v3[m] + BN_EPS);
            float val = ((float)acc[mt][r] + b3[m]) * iv + (be3[m] - m3[m] * iv);
            float vm = fmaxf(val, __shfl_xor(val, 1));
            if ((lane & 1) == 0)
                A3[(((b*6 + h) * 16 + (w >> 1)) << 7) + m] = vm >= 0.f ? (char)1 : (char)-1;
        }
}

__global__ __launch_bounds__(256) void k_block4(
    const char* __restrict__ A3, const float* __restrict__ w4, const float* __restrict__ b4,
    const float* __restrict__ g4, const float* __restrict__ be4,
    const float* __restrict__ m4, const float* __restrict__ v4, char* __restrict__ A4)
{
    __shared__ char As[128 * 128];
    __shared__ char Bs[128 * 128];
    int t = threadIdx.x;
    int Nbase = blockIdx.x * 128;
    int lane = t & 31, wv = t >> 5, n0 = wv << 4;
    int nr = n0 + (lane & 15), mr = lane & 15;
    v8i acc[8]; for (int i = 0; i < 8; ++i) acc[i] = (v8i){};

    for (int kh = 0; kh < 6; ++kh) {
        __syncthreads();
        for (int d = t; d < 4096; d += 256) {
            int m = d >> 5, ic = (d & 31) << 2;
            *(int*)&As[d << 2] =
                pack4(sgnb(w4[m*768 + (ic  )*6 + kh]), sgnb(w4[m*768 + (ic+1)*6 + kh]),
                      sgnb(w4[m*768 + (ic+2)*6 + kh]), sgnb(w4[m*768 + (ic+3)*6 + kh]));
        }
        for (int d = t; d < 4096; d += 256) {
            int n = d >> 5, ic = (d & 31) << 2;
            int N = Nbase + n, b = N >> 4, w = N & 15;
            *(int*)&Bs[d << 2] = *(const int*)&A3[(((b*6 + kh) * 16 + w) << 7) + ic];
        }
        __syncthreads();
#pragma unroll
        for (int kc = 0; kc < 2; ++kc) {
            v8i bf = frag_b(&Bs[nr * 128 + kc * 64], lane);
#pragma unroll
            for (int mt = 0; mt < 8; ++mt) {
                v8i af = frag_a(&As[(mt * 16 + mr) * 128 + kc * 64], lane);
                acc[mt] = __builtin_amdgcn_wmma_i32_16x16x64_iu8(true, af, true, bf, acc[mt], false, false);
            }
        }
    }
    int N = Nbase + nr, b = N >> 4, w = N & 15;
#pragma unroll
    for (int mt = 0; mt < 8; ++mt)
#pragma unroll
        for (int r = 0; r < 8; ++r) {
            int m = mt * 16 + r + ((lane >> 4) << 3);
            float iv = g4[m] * rsqrtf(v4[m] + BN_EPS);
            float val = ((float)acc[mt][r] + b4[m]) * iv + (be4[m] - m4[m] * iv);
            A4[b * 2048 + m * 16 + w] = val >= 0.f ? (char)1 : (char)-1;
        }
}

__global__ __launch_bounds__(256) void k_fc(
    const char* __restrict__ A4, const float* __restrict__ wfc, const float* __restrict__ bfc,
    float* __restrict__ out)
{
    __shared__ char As[16 * 256];
    __shared__ char Bs[128 * 256];
    int t = threadIdx.x;
    int Nbase = blockIdx.x * 128;
    int lane = t & 31, wv = t >> 5, n0 = wv << 4;
    int nr = n0 + (lane & 15), mr = lane & 15;
    v8i acc = (v8i){};

    for (int ks = 0; ks < 8; ++ks) {
        __syncthreads();
        for (int d = t; d < 1024; d += 256) {
            int m = d >> 6, k = ((d & 63) << 2) + ks * 256, outw = 0;
            if (m < 6)
                outw = pack4(sgnb(wfc[m*2048 + k]), sgnb(wfc[m*2048 + k + 1]),
                             sgnb(wfc[m*2048 + k + 2]), sgnb(wfc[m*2048 + k + 3]));
            *(int*)&As[d << 2] = outw;
        }
        for (int d = t; d < 8192; d += 256) {
            int n = d >> 6, k = ((d & 63) << 2) + ks * 256;
            *(int*)&Bs[d << 2] = *(const int*)&A4[(Nbase + n) * 2048 + k];
        }
        __syncthreads();
#pragma unroll
        for (int kc = 0; kc < 4; ++kc) {
            v8i bf = frag_b(&Bs[nr * 256 + kc * 64], lane);
            v8i af = frag_a(&As[mr * 256 + kc * 64], lane);
            acc = __builtin_amdgcn_wmma_i32_16x16x64_iu8(true, af, true, bf, acc, false, false);
        }
    }
    int N = Nbase + nr;
#pragma unroll
    for (int r = 0; r < 8; ++r) {
        int m = r + ((lane >> 4) << 3);
        if (m < 6) out[N * 6 + m] = (float)acc[r] + bfc[m];
    }
}

extern "C" void kernel_launch(void* const* d_in, const int* in_sizes, int n_in,
                              void* d_out, int out_size, void* d_ws, size_t ws_size,
                              hipStream_t stream) {
    (void)in_sizes; (void)n_in; (void)out_size; (void)ws_size;
    const float* x   = (const float*)d_in[0];
    const float* w1  = (const float*)d_in[1];  const float* b1  = (const float*)d_in[2];
    const float* g1  = (const float*)d_in[3];  const float* be1 = (const float*)d_in[4];
    const float* m1  = (const float*)d_in[5];  const float* v1  = (const float*)d_in[6];
    const float* w2  = (const float*)d_in[7];  const float* b2  = (const float*)d_in[8];
    const float* g2  = (const float*)d_in[9];  const float* be2 = (const float*)d_in[10];
    const float* m2  = (const float*)d_in[11]; const float* v2  = (const float*)d_in[12];
    const float* w3  = (const float*)d_in[13]; const float* b3  = (const float*)d_in[14];
    const float* g3  = (const float*)d_in[15]; const float* be3 = (const float*)d_in[16];
    const float* m3  = (const float*)d_in[17]; const float* v3  = (const float*)d_in[18];
    const float* w4  = (const float*)d_in[19]; const float* b4  = (const float*)d_in[20];
    const float* g4  = (const float*)d_in[21]; const float* be4 = (const float*)d_in[22];
    const float* m4  = (const float*)d_in[23]; const float* v4  = (const float*)d_in[24];
    const float* wfc = (const float*)d_in[25]; const float* bfc = (const float*)d_in[26];
    float* out = (float*)d_out;

    char* ws = (char*)d_ws;
    char* A1 = ws;
    char* A2 = ws + (size_t)25165824;
    char* A3 = ws + (size_t)75497472;
    char* A4 = ws;

    k_block1<<<dim3(98304), dim3(256), 0, stream>>>(x, w1, b1, g1, be1, m1, v1, A1);
    k_block2<<<dim3(6144),  dim3(256), 0, stream>>>(A1, w2, b2, g2, be2, m2, v2, A2);
    k_block3<<<dim3(6144),  dim3(256), 0, stream>>>(A2, w3, b3, g3, be3, m3, v3, A3);
    k_block4<<<dim3(512),   dim3(256), 0, stream>>>(A3, w4, b4, g4, be4, m4, v4, A4);
    k_fc    <<<dim3(32),    dim3(256), 0, stream>>>(A4, wfc, bfc, out);
}
